// CustomMultiHeadSelfAttention_87806311399970
// MI455X (gfx1250) — matrix operand order repaired and checked by reading, not compiled, not run on hardware
//
#include <hip/hip_runtime.h>

// ---------------------------------------------------------------------------
// Types matching CDNA5 WMMA builtin signatures
// ---------------------------------------------------------------------------
typedef __attribute__((ext_vector_type(16))) __bf16 v16bf;
typedef __attribute__((ext_vector_type(8)))  __bf16 v8bf;
typedef __attribute__((ext_vector_type(8)))  float  v8f;

union FragU { v16bf v; v8bf h[2]; };

#if __has_builtin(__builtin_amdgcn_global_load_async_to_lds_b128) && \
    __has_builtin(__builtin_amdgcn_s_wait_asynccnt)
#define HAS_ASYNC_LDS 1
#else
#define HAS_ASYNC_LDS 0
#endif

#define GLOBAL_AS __attribute__((address_space(1)))
#define LDS_AS    __attribute__((address_space(3)))

// 128-bit payload type the async builtin expects (gcc-style vector of 4 ints)
typedef int v4i_pay __attribute__((vector_size(16)));

#if HAS_ASYNC_LDS
__device__ __attribute__((always_inline)) inline void
async_cp_b128(const void* g, void* l) {
    __builtin_amdgcn_global_load_async_to_lds_b128(
        (GLOBAL_AS v4i_pay*)(g), (LDS_AS v4i_pay*)(l), 0, 0);
}
#endif

__device__ __attribute__((always_inline)) inline v8f v8f_zero() {
    v8f z;
#pragma unroll
    for (int i = 0; i < 8; ++i) z[i] = 0.0f;
    return z;
}

// A-fragment (16x32, bf16) from row-major src (ld elements per row).
// ISA 7.12.2: lane(half,m): VGPR0-3 = K(half*8 .. +7), VGPR4-7 = K(16+half*8 .. +7)
__device__ __attribute__((always_inline)) inline v16bf
load_frag_a(const __bf16* src, int ld, int lane) {
    const int hl = (lane >> 4) & 1;
    const int m  = lane & 15;
    const __bf16* p = src + (size_t)m * ld + hl * 8;
    FragU f;
    f.h[0] = *(const v8bf*)(p);
    f.h[1] = *(const v8bf*)(p + 16);
    return f.v;
}

// B-fragment (32x16, bf16) where operand is stored W-style row-major [N][K]:
// B[k][n] = W[n][k].  lane(half,n): K(half*8 .. +7), then K(16+half*8 .. +7).
__device__ __attribute__((always_inline)) inline v16bf
load_frag_b(const __bf16* src, int ld, int lane) {
    const int hl = (lane >> 4) & 1;
    const int n  = lane & 15;
    const __bf16* p = src + (size_t)n * ld + hl * 8;
    FragU f;
    f.h[0] = *(const v8bf*)(p);
    f.h[1] = *(const v8bf*)(p + 16);
    return f.v;
}

// ---------------------------------------------------------------------------
// f32 -> bf16 conversion
// ---------------------------------------------------------------------------
__global__ void f32_to_bf16_kernel(const float* __restrict__ src,
                                   __bf16* __restrict__ dst, size_t n) {
    size_t i = (size_t)blockIdx.x * blockDim.x + threadIdx.x;
    size_t stride = (size_t)gridDim.x * blockDim.x;
    for (; i < n; i += stride) dst[i] = (__bf16)src[i];
}

// ---------------------------------------------------------------------------
// GEMM: C[M,Nout] = A[M,K](bf16) @ W[Nout,K](bf16)^T + bias
// Block tile 128x256, 256 threads = 8 waves (2Mx4N), wave tile 64x64:
// 16 x v_wmma_f32_16x16x32_bf16 per K-step of 32 against 16 ds_load_b128.
// Double-buffered LDS, one barrier per K-step; staging uses the gfx1250
// async-to-LDS path (ASYNCcnt) when the toolchain exposes it.
// mode 0: store f32 row-major to outF.
// mode 1: scatter bf16 into q/k/v buffers laid out [B,H,N,64].
// ---------------------------------------------------------------------------
__global__ __launch_bounds__(256) void gemm_bf16_kernel(
    const __bf16* __restrict__ A, const __bf16* __restrict__ W,
    const float* __restrict__ bias, float* __restrict__ outF,
    __bf16* __restrict__ qb, __bf16* __restrict__ kb, __bf16* __restrict__ vb,
    int M, int K, int Nout, int mode)
{
    constexpr int BM = 128, BN = 256, BK = 32;
    __shared__ __bf16 As[2][BM * BK];   // 2 x 8 KB
    __shared__ __bf16 Ws[2][BN * BK];   // 2 x 16 KB

    const int tid  = threadIdx.x;
    const int lane = tid & 31;
    const int wid  = tid >> 5;
    const int wm   = wid >> 2;   // 0..1 : 64 rows
    const int wn   = wid & 3;    // 0..3 : 64 cols
    const int m_base = blockIdx.y * BM;
    const int n_base = blockIdx.x * BN;

    const __bf16* Ab = A + (size_t)m_base * K;
    const __bf16* Wb = W + (size_t)n_base * K;

    // per-thread staging chunks (8 bf16 = 16 B each):
    // A tile: 512 chunks -> 2/thread; W tile: 1024 chunks -> 4/thread
    const int arow[2] = { (tid + 0)   >> 2, (tid + 256) >> 2 };
    const int acol    = (tid & 3) * 8;
    int wrow[4];
#pragma unroll
    for (int c = 0; c < 4; ++c) wrow[c] = (tid + c * 256) >> 2;
    const int wcol = (tid & 3) * 8;

#if HAS_ASYNC_LDS
    auto stage = [&](int k0, __bf16* dA, __bf16* dW) {
#pragma unroll
        for (int c = 0; c < 2; ++c)
            async_cp_b128(&Ab[(size_t)arow[c] * K + k0 + acol],
                          &dA[arow[c] * BK + acol]);
#pragma unroll
        for (int c = 0; c < 4; ++c)
            async_cp_b128(&Wb[(size_t)wrow[c] * K + k0 + wcol],
                          &dW[wrow[c] * BK + wcol]);
    };
#endif

    v8f acc[4][4];
#pragma unroll
    for (int mt = 0; mt < 4; ++mt)
#pragma unroll
        for (int nt = 0; nt < 4; ++nt) acc[mt][nt] = v8f_zero();

    // ---- stage first tile into buffer 0 ----
#if HAS_ASYNC_LDS
    stage(0, As[0], Ws[0]);
    __builtin_amdgcn_s_wait_asynccnt(0);
#else
    {
        v8bf ar[2], wr[4];
#pragma unroll
        for (int c = 0; c < 2; ++c)
            ar[c] = *(const v8bf*)&Ab[(size_t)arow[c] * K + acol];
#pragma unroll
        for (int c = 0; c < 4; ++c)
            wr[c] = *(const v8bf*)&Wb[(size_t)wrow[c] * K + wcol];
#pragma unroll
        for (int c = 0; c < 2; ++c)
            *(v8bf*)&As[0][arow[c] * BK + acol] = ar[c];
#pragma unroll
        for (int c = 0; c < 4; ++c)
            *(v8bf*)&Ws[0][wrow[c] * BK + wcol] = wr[c];
    }
#endif
    __syncthreads();

    for (int k0 = 0; k0 < K; k0 += BK) {
        const int cur = (k0 >> 5) & 1;
        const bool has_next = (k0 + BK) < K;

#if HAS_ASYNC_LDS
        if (has_next) stage(k0 + BK, As[cur ^ 1], Ws[cur ^ 1]);
#else
        v8bf ar[2], wr[4];
        if (has_next) {
            const int kn = k0 + BK;
#pragma unroll
            for (int c = 0; c < 2; ++c)
                ar[c] = *(const v8bf*)&Ab[(size_t)arow[c] * K + kn + acol];
#pragma unroll
            for (int c = 0; c < 4; ++c)
                wr[c] = *(const v8bf*)&Wb[(size_t)wrow[c] * K + kn + wcol];
        }
#endif
        // ---- compute on current buffer: 16 WMMAs ----
        v16bf bfr[4];
#pragma unroll
        for (int nt = 0; nt < 4; ++nt)
            bfr[nt] = load_frag_b(&Ws[cur][(wn * 64 + nt * 16) * BK], BK, lane);
#pragma unroll
        for (int mt = 0; mt < 4; ++mt) {
            v16bf af = load_frag_a(&As[cur][(wm * 64 + mt * 16) * BK], BK, lane);
#pragma unroll
            for (int nt = 0; nt < 4; ++nt)
                acc[mt][nt] = __builtin_amdgcn_wmma_f32_16x16x32_bf16(
                    false, af, false, bfr[nt], (short)0, acc[mt][nt],
                    false, false);
        }

#if HAS_ASYNC_LDS
        __builtin_amdgcn_s_wait_asynccnt(0);
#else
        if (has_next) {
#pragma unroll
            for (int c = 0; c < 2; ++c)
                *(v8bf*)&As[cur ^ 1][arow[c] * BK + acol] = ar[c];
#pragma unroll
            for (int c = 0; c < 4; ++c)
                *(v8bf*)&Ws[cur ^ 1][wrow[c] * BK + wcol] = wr[c];
        }
#endif
        __syncthreads();
    }

    // Epilogue: C/D layout lane(half,n), VGPR r <-> row half*8+r
    const int hl = lane >> 4;
    const int nn = lane & 15;
#pragma unroll
    for (int mt = 0; mt < 4; ++mt) {
#pragma unroll
        for (int nt = 0; nt < 4; ++nt) {
#pragma unroll
            for (int r = 0; r < 8; ++r) {
                const int row = m_base + wm * 64 + mt * 16 + hl * 8 + r;
                const int col = n_base + wn * 64 + nt * 16 + nn;
                const float val = acc[mt][nt][r] + bias[col];
                if (mode == 0) {
                    outF[(size_t)row * Nout + col] = val;
                } else {
                    const int which = col >> 10;        // 0:q 1:k 2:v
                    const int e = col & 1023;
                    const int h = e >> 6, d = e & 63;
                    const int b = row >> 11, nq = row & 2047;
                    const size_t off =
                        (((size_t)b * 16 + h) * 2048 + nq) * 64 + d;
                    const __bf16 bv = (__bf16)val;
                    if (which == 0)      qb[off] = bv;
                    else if (which == 1) kb[off] = bv;
                    else                 vb[off] = bv;
                }
            }
        }
    }
}

// ---------------------------------------------------------------------------
// Flash attention: one block per (b*h, 64-query-row slab); 4 waves, each wave
// owns 16 query rows.  Online softmax, all matmuls via WMMA bf16.
// Q,K,V in [B,H,N,64] bf16; output written bf16 into [B,N,E] (heads re-packed).
// ---------------------------------------------------------------------------
__global__ __launch_bounds__(128) void attn_kernel(
    const __bf16* __restrict__ Q, const __bf16* __restrict__ Kx,
    const __bf16* __restrict__ V, __bf16* __restrict__ O, int N)
{
    __shared__ __bf16 Ks[32 * 64];      // K block, [j][d]
    __shared__ __bf16 Vs[64 * 32];      // V block transposed, [d][j]
    __shared__ __bf16 Ps[4][16 * 32];   // per-wave P tile

    const int tid  = threadIdx.x;
    const int lane = tid & 31;
    const int wid  = tid >> 5;
    const int bh   = blockIdx.x;             // 0..B*H-1
    const int b    = bh >> 4, h = bh & 15;
    const int q0   = blockIdx.y * 64 + wid * 16;
    const size_t head_off = (size_t)bh * N * 64;
    const __bf16* Qh = Q  + head_off;
    const __bf16* Kh = Kx + head_off;
    const __bf16* Vh = V  + head_off;

    // Q fragments held in registers for the whole pass (16x64 = 2 A-frags)
    v16bf qf[2];
    qf[0] = load_frag_a(Qh + (size_t)q0 * 64 + 0,  64, lane);
    qf[1] = load_frag_a(Qh + (size_t)q0 * 64 + 32, 64, lane);

    v8f o[4];
#pragma unroll
    for (int t = 0; t < 4; ++t) o[t] = v8f_zero();
    float mrow[8], lrow[8];
#pragma unroll
    for (int r = 0; r < 8; ++r) { mrow[r] = -1e30f; lrow[r] = 0.0f; }

    const int hl = lane >> 4;
    const int nn = lane & 15;

    for (int j0 = 0; j0 < N; j0 += 32) {
        __syncthreads();
        // stage K rows (async path when available); V rows transposed to [d][j]
#pragma unroll
        for (int i = 0; i < 2; ++i) {
            const int chunk = tid * 2 + i;       // 0..255 chunks of 8 bf16
            const int row   = chunk >> 3;        // 0..31
            const int cc    = (chunk & 7) * 8;   // 0..56
#if HAS_ASYNC_LDS
            async_cp_b128(&Kh[(size_t)(j0 + row) * 64 + cc],
                          &Ks[row * 64 + cc]);
#else
            *(v8bf*)&Ks[row * 64 + cc] =
                *(const v8bf*)&Kh[(size_t)(j0 + row) * 64 + cc];
#endif
            v8bf vv = *(const v8bf*)&Vh[(size_t)(j0 + row) * 64 + cc];
#pragma unroll
            for (int e = 0; e < 8; ++e) Vs[(cc + e) * 32 + row] = vv[e];
        }
#if HAS_ASYNC_LDS
        __builtin_amdgcn_s_wait_asynccnt(0);
#endif
        __syncthreads();

        // scores: S(16x32) = Q(16x64) . K(32x64)^T, two 16x16 WMMA column tiles
        v8f s[2];
#pragma unroll
        for (int jt = 0; jt < 2; ++jt) {
            s[jt] = v8f_zero();
            s[jt] = __builtin_amdgcn_wmma_f32_16x16x32_bf16(
                false, qf[0], false,
                load_frag_b(&Ks[(jt * 16) * 64 + 0], 64, lane),
                (short)0, s[jt], false, false);
            s[jt] = __builtin_amdgcn_wmma_f32_16x16x32_bf16(
                false, qf[1], false,
                load_frag_b(&Ks[(jt * 16) * 64 + 32], 64, lane),
                (short)0, s[jt], false, false);
        }

        // online softmax (rows half*8+r, columns across 16 lanes of the half)
#pragma unroll
        for (int r = 0; r < 8; ++r) {
            float s0 = s[0][r] * 0.125f;   // 1/sqrt(64)
            float s1 = s[1][r] * 0.125f;
            float cm = fmaxf(s0, s1);
#pragma unroll
            for (int msk = 1; msk < 16; msk <<= 1)
                cm = fmaxf(cm, __shfl_xor(cm, msk, 32));
            const float mn    = fmaxf(mrow[r], cm);
            const float alpha = __expf(mrow[r] - mn);
            mrow[r] = mn;
            const float p0 = __expf(s0 - mn);
            const float p1 = __expf(s1 - mn);
            s[0][r] = p0; s[1][r] = p1;
            float rs = p0 + p1;
#pragma unroll
            for (int msk = 1; msk < 16; msk <<= 1)
                rs += __shfl_xor(rs, msk, 32);
            lrow[r] = lrow[r] * alpha + rs;
#pragma unroll
            for (int t = 0; t < 4; ++t) o[t][r] *= alpha;
        }

        // re-shape P (C-layout f32) -> A-fragment (bf16) via per-wave LDS tile
        __bf16* Pw = Ps[wid];
#pragma unroll
        for (int r = 0; r < 8; ++r) {
            Pw[(hl * 8 + r) * 32 + nn]      = (__bf16)s[0][r];
            Pw[(hl * 8 + r) * 32 + 16 + nn] = (__bf16)s[1][r];
        }
        v16bf pf = load_frag_a(Pw, 32, lane);   // same-wave LDS ops are in-order

        // O(16x64) += P(16x32) . V(32x64); Vs is [d][j] so B-frags are rows
#pragma unroll
        for (int t = 0; t < 4; ++t)
            o[t] = __builtin_amdgcn_wmma_f32_16x16x32_bf16(
                false, pf, false,
                load_frag_b(&Vs[(t * 16) * 32], 32, lane),
                (short)0, o[t], false, false);
    }

    // normalize and write bf16 into [B,N,E] with heads repacked at h*64
#pragma unroll
    for (int r = 0; r < 8; ++r) lrow[r] = 1.0f / lrow[r];
#pragma unroll
    for (int t = 0; t < 4; ++t) {
#pragma unroll
        for (int r = 0; r < 8; ++r) {
            const size_t row = (size_t)b * 2048 + (q0 + hl * 8 + r);
            O[row * 1024 + h * 64 + t * 16 + nn] = (__bf16)(o[t][r] * lrow[r]);
        }
    }
}

// ---------------------------------------------------------------------------
// Host launcher
// ---------------------------------------------------------------------------
extern "C" void kernel_launch(void* const* d_in, const int* in_sizes, int n_in,
                              void* d_out, int out_size, void* d_ws, size_t ws_size,
                              hipStream_t stream) {
    (void)in_sizes; (void)n_in; (void)out_size; (void)ws_size;
    constexpr int B = 4, N = 2048, E = 1024, H = 16;
    const float* qkv   = (const float*)d_in[0];
    const float* in_w  = (const float*)d_in[1];
    const float* in_b  = (const float*)d_in[2];
    const float* out_w = (const float*)d_in[3];
    const float* out_b = (const float*)d_in[4];
    float* out = (float*)d_out;

    const size_t ME = (size_t)B * N * E;          // 8,388,608
    char* ws = (char*)d_ws;
    __bf16* x_bf  = (__bf16*)ws; ws += ME * 2;                    // activations
    __bf16* wi_bf = (__bf16*)ws; ws += (size_t)3 * E * E * 2;     // in_proj_w
    __bf16* wo_bf = (__bf16*)ws; ws += (size_t)E * E * 2;         // out_proj_w
    __bf16* qb    = (__bf16*)ws; ws += ME * 2;
    __bf16* kb    = (__bf16*)ws; ws += ME * 2;
    __bf16* vb    = (__bf16*)ws; ws += ME * 2;
    __bf16* ab    = (__bf16*)ws; ws += ME * 2;                    // attn out

    f32_to_bf16_kernel<<<1024, 256, 0, stream>>>(qkv,   x_bf,  ME);
    f32_to_bf16_kernel<<<1024, 256, 0, stream>>>(in_w,  wi_bf, (size_t)3 * E * E);
    f32_to_bf16_kernel<<<1024, 256, 0, stream>>>(out_w, wo_bf, (size_t)E * E);

    // in-projection: [8192,1024] x [3072,1024]^T -> scatter q/k/v [B,H,N,64]
    dim3 g1(3 * E / 256, (B * N) / 128);   // (12, 64)
    gemm_bf16_kernel<<<g1, 256, 0, stream>>>(x_bf, wi_bf, in_b, nullptr,
                                             qb, kb, vb, B * N, E, 3 * E, 1);

    // attention
    dim3 g2(B * H, N / 64);                // (64, 32)
    attn_kernel<<<g2, 128, 0, stream>>>(qb, kb, vb, ab, N);

    // out-projection: [8192,1024] x [1024,1024]^T -> f32 d_out
    dim3 g3(E / 256, (B * N) / 128);       // (4, 64)
    gemm_bf16_kernel<<<g3, 256, 0, stream>>>(ab, wo_bf, out_b, out,
                                             nullptr, nullptr, nullptr,
                                             B * N, E, E, 0);
}
